// MemoryReader_60670708023447
// MI455X (gfx1250) — matrix operand order repaired and checked by reading, not compiled, not run on hardware
//
#include <hip/hip_runtime.h>

typedef __attribute__((ext_vector_type(16))) __bf16 v16bf;
typedef __attribute__((ext_vector_type(8)))  __bf16 bfx8;
typedef __attribute__((ext_vector_type(2)))  __bf16 bfx2;
typedef __attribute__((ext_vector_type(8)))  float  v8f;

#define NPOS 3136
#define NB   16
#define CK_  64
#define CV_  512
#define QB   64
#define KB   64
#define KSTR 72   // padded LDS row stride in bf16 units (144B, 16B aligned, conflict-free)

__device__ __forceinline__ v16bf cat16(bfx8 a, bfx8 b) {
  return __builtin_shufflevector(a, b, 0,1,2,3,4,5,6,7,8,9,10,11,12,13,14,15);
}
__device__ __forceinline__ bfx2 pack_bf16(float a, float b) {
#if __has_builtin(__builtin_amdgcn_cvt_pk_bf16_f32)
  return __builtin_bit_cast(bfx2, __builtin_amdgcn_cvt_pk_bf16_f32(a, b));
#else
  bfx2 r; r.x = (__bf16)a; r.y = (__bf16)b; return r;
#endif
}
__device__ __forceinline__ float hmax16(float v) {
  v = fmaxf(v, __shfl_xor(v, 1));
  v = fmaxf(v, __shfl_xor(v, 2));
  v = fmaxf(v, __shfl_xor(v, 4));
  v = fmaxf(v, __shfl_xor(v, 8));
  return v;
}
__device__ __forceinline__ float hsum16(float v) {
  v += __shfl_xor(v, 1);
  v += __shfl_xor(v, 2);
  v += __shfl_xor(v, 4);
  v += __shfl_xor(v, 8);
  return v;
}

__global__ __launch_bounds__(256)
void fa_readout_kernel(const float* __restrict__ mk, const float* __restrict__ qk,
                       const float* __restrict__ mv, float* __restrict__ out) {
  __shared__ __bf16 sV[CV_ * KSTR];       // 73728 B  V tile [cv][key]; reused as fp32 out-stage
  __shared__ __bf16 sK[KB * KSTR];        //  9216 B  K tile [key][ck]
  __shared__ __bf16 sQ[QB * KSTR];        //  9216 B  Q tile [q][ck]
  __shared__ __bf16 sP[8 * 16 * KSTR];    // 18432 B  per-wave P tiles [q][key]
  __shared__ float  sBiasP[4][KB];        //  1024 B  per-key sum-of-squares partials

  const int tid    = threadIdx.x;
  const int lane   = tid & 31;
  const int wv     = tid >> 5;
  const int h      = wv & 1;      // which 256-wide CV half
  const int g      = wv >> 1;     // which 16-query group
  const int laneLo = lane & 15;
  const int hiHalf = lane >> 4;

  const int b  = blockIdx.y;
  const int q0 = blockIdx.x * QB;
  const size_t bK = (size_t)b * CK_ * NPOS;
  const size_t bV = (size_t)b * CV_ * NPOS;

  // ---- stage Q (fp32 -> bf16 pairs), row-major [q][ck] ----
  {
    const int qq = tid & 63, cb = tid >> 6;
    #pragma unroll
    for (int j = 0; j < 8; ++j) {
      int c = cb * 16 + 2 * j;
      float a0 = qk[bK + (size_t)c * NPOS + q0 + qq];
      float a1 = qk[bK + (size_t)(c + 1) * NPOS + q0 + qq];
      *(bfx2*)&sQ[qq * KSTR + c] = pack_bf16(a0, a1);
    }
  }
  __syncthreads();

  // hoist the two Q A-operands (ck k-steps 0..31, 32..63); A-layout:
  // lanes 0-15: M=lane, K = s*32 + {0..15}; lanes 16-31: K = s*32 + {16..31}
  v16bf Qa[2];
  {
    const __bf16* qrow = &sQ[(g * 16 + laneLo) * KSTR + hiHalf * 16];
    #pragma unroll
    for (int s = 0; s < 2; ++s)
      Qa[s] = cat16(*(const bfx8*)(qrow + s * 32), *(const bfx8*)(qrow + s * 32 + 8));
  }

  v8f Oacc[16];
  #pragma unroll
  for (int j = 0; j < 16; ++j) Oacc[j] = (v8f)(0.0f);
  float mrow[8], lrow[8];
  #pragma unroll
  for (int r = 0; r < 8; ++r) { mrow[r] = -3.0e38f; lrow[r] = 0.0f; }

  __bf16* sPw = &sP[wv * 16 * KSTR];

  for (int it = 0; it < NPOS / KB; ++it) {
    const int k0 = it * KB;
    // ---- stage K tile transposed to [key][ck] + per-key sum-of-squares ----
    {
      const int kk = tid & 63, cb = tid >> 6;
      float ss = 0.0f;
      #pragma unroll
      for (int j = 0; j < 8; ++j) {
        int c = cb * 16 + 2 * j;
        float a0 = mk[bK + (size_t)c * NPOS + k0 + kk];
        float a1 = mk[bK + (size_t)(c + 1) * NPOS + k0 + kk];
        ss += a0 * a0 + a1 * a1;
        *(bfx2*)&sK[kk * KSTR + c] = pack_bf16(a0, a1);
      }
      sBiasP[cb][kk] = ss;
    }
    // ---- stage V tile [cv][key] ----
    {
      const int kp = tid & 31, cr = tid >> 5;
      #pragma unroll 8
      for (int j = 0; j < 64; ++j) {
        int c = j * 8 + cr;
        float2 v2 = *(const float2*)(mv + bV + (size_t)c * NPOS + k0 + kp * 2);
        *(bfx2*)&sV[c * KSTR + kp * 2] = pack_bf16(v2.x, v2.y);
      }
    }
    __syncthreads();

    // per-lane bias (log2 domain): ||k||^2 * 0.125 * log2(e) for this lane's 4 key columns
    float b2[4];
    #pragma unroll
    for (int j = 0; j < 4; ++j) {
      int key = j * 16 + laneLo;
      b2[j] = (sBiasP[0][key] + sBiasP[1][key] + sBiasP[2][key] + sBiasP[3][key]) * 0.18033688f;
    }

    // ---- GEMM1: S[16q x 64key] = Q x K^T  (depth-2 pipelined B loads) ----
    v8f Sacc[4];
    {
      const __bf16* krow = &sK[laneLo * KSTR + hiHalf * 16];
      v16bf B0 = cat16(*(const bfx8*)(krow + 0),  *(const bfx8*)(krow + 8));
      v16bf B1 = cat16(*(const bfx8*)(krow + 32), *(const bfx8*)(krow + 40));
      #pragma unroll
      for (int j = 0; j < 4; ++j) {
        v16bf C0 = B0, C1 = B1;
        if (j < 3) {
          const __bf16* nr = krow + (j + 1) * 16 * KSTR;
          B0 = cat16(*(const bfx8*)(nr + 0),  *(const bfx8*)(nr + 8));
          B1 = cat16(*(const bfx8*)(nr + 32), *(const bfx8*)(nr + 40));
        }
        v8f c = (v8f)(0.0f);
        c = __builtin_amdgcn_wmma_f32_16x16x32_bf16(false, Qa[0], false, C0, (short)0, c, false, false);
        c = __builtin_amdgcn_wmma_f32_16x16x32_bf16(false, Qa[1], false, C1, (short)0, c, false, false);
        Sacc[j] = c;
      }
    }

    // ---- online softmax (log2 domain): logit*log2e = Sraw*0.25*log2e - bias2 ----
    float t[4][8];
    #pragma unroll
    for (int j = 0; j < 4; ++j) {
      #pragma unroll
      for (int r = 0; r < 8; ++r)
        t[j][r] = Sacc[j][r] * 0.36067376f - b2[j];
    }
    float corr[8];
    float cmin = 1.0f;
    #pragma unroll
    for (int r = 0; r < 8; ++r) {
      float mx = fmaxf(fmaxf(t[0][r], t[1][r]), fmaxf(t[2][r], t[3][r]));
      mx = hmax16(mx);
      float mnew = fmaxf(mrow[r], mx);
      corr[r] = __builtin_amdgcn_exp2f(mrow[r] - mnew);
      mrow[r] = mnew;
      cmin = fminf(cmin, corr[r]);
      float ls = 0.0f;
      #pragma unroll
      for (int j = 0; j < 4; ++j) {
        float p = __builtin_amdgcn_exp2f(t[j][r] - mnew);
        t[j][r] = p;
        ls += p;
      }
      lrow[r] = lrow[r] * corr[r] + ls;   // per-lane partial; reduced at the end
    }
    // rescale accumulator only when some row max actually moved (wave-uniform branch)
    cmin = fminf(cmin, __shfl_xor(cmin, 16));
    if (cmin < 1.0f) {
      #pragma unroll
      for (int r = 0; r < 8; ++r) {
        #pragma unroll
        for (int j = 0; j < 16; ++j) Oacc[j][r] *= corr[r];
      }
    }

    // ---- P (bf16) -> per-wave LDS, C-layout positions ----
    #pragma unroll
    for (int r = 0; r < 8; ++r) {
      #pragma unroll
      for (int j = 0; j < 4; ++j)
        sPw[(r + 8 * hiHalf) * KSTR + j * 16 + laneLo] = (__bf16)t[j][r];
    }
    asm volatile("s_wait_dscnt 0" ::: "memory");   // cross-lane P visible before A-load

    // ---- GEMM2: Oacc[16q x 256cv] += P x V  (depth-2 pipelined B loads) ----
    v16bf Pa[2];
    {
      const __bf16* prow = &sPw[laneLo * KSTR + hiHalf * 16];
      #pragma unroll
      for (int s = 0; s < 2; ++s)
        Pa[s] = cat16(*(const bfx8*)(prow + s * 32), *(const bfx8*)(prow + s * 32 + 8));
    }
    {
      const __bf16* vrow = &sV[(h * 256 + laneLo) * KSTR + hiHalf * 16];
      v16bf B0 = cat16(*(const bfx8*)(vrow + 0),  *(const bfx8*)(vrow + 8));
      v16bf B1 = cat16(*(const bfx8*)(vrow + 32), *(const bfx8*)(vrow + 40));
      #pragma unroll
      for (int j = 0; j < 16; ++j) {
        v16bf C0 = B0, C1 = B1;
        if (j < 15) {
          const __bf16* nr = vrow + (j + 1) * 16 * KSTR;
          B0 = cat16(*(const bfx8*)(nr + 0),  *(const bfx8*)(nr + 8));
          B1 = cat16(*(const bfx8*)(nr + 32), *(const bfx8*)(nr + 40));
        }
        Oacc[j] = __builtin_amdgcn_wmma_f32_16x16x32_bf16(false, Pa[0], false, C0,
                                                          (short)0, Oacc[j], false, false);
        Oacc[j] = __builtin_amdgcn_wmma_f32_16x16x32_bf16(false, Pa[1], false, C1,
                                                          (short)0, Oacc[j], false, false);
      }
    }
    __syncthreads();   // compute done before next iteration overwrites tiles
  }

  // ---- finalize: divide by full row sums ----
  float rinv[8];
  #pragma unroll
  for (int r = 0; r < 8; ++r) rinv[r] = __builtin_amdgcn_rcpf(hsum16(lrow[r]));
  #pragma unroll
  for (int j = 0; j < 16; ++j) {
    #pragma unroll
    for (int r = 0; r < 8; ++r) Oacc[j][r] *= rinv[r];
  }

  // ---- coalesced output via LDS bounce (reuse sV as fp32 [256cv][66q]) ----
  float* ob = (float*)sV;
  #pragma unroll 1
  for (int ch = 0; ch < 2; ++ch) {
    if (h == ch) {
      #pragma unroll
      for (int j = 0; j < 16; ++j) {
        int cvr = j * 16 + laneLo;
        #pragma unroll
        for (int r = 0; r < 8; ++r)
          ob[cvr * 66 + (g * 16 + r + 8 * hiHalf)] = Oacc[j][r];
      }
    }
    __syncthreads();
    {
      const int q = tid & 63, cb = tid >> 6;
      #pragma unroll 8
      for (int i = 0; i < 64; ++i) {
        int cvr = cb + i * 4;
        out[bV + (size_t)(ch * 256 + cvr) * NPOS + q0 + q] = ob[cvr * 66 + q];
      }
    }
    __syncthreads();
  }
}

__global__ void qv_copy_kernel(const float4* __restrict__ src, float4* __restrict__ dst, int n4) {
  int i = blockIdx.x * blockDim.x + threadIdx.x;
  if (i < n4) dst[i] = src[i];
}

extern "C" void kernel_launch(void* const* d_in, const int* in_sizes, int n_in,
                              void* d_out, int out_size, void* d_ws, size_t ws_size,
                              hipStream_t stream) {
  const float* mk = (const float*)d_in[0];
  const float* qk = (const float*)d_in[1];
  const float* mv = (const float*)d_in[2];
  const float* qv = (const float*)d_in[3];
  float* out = (float*)d_out;

  dim3 grid(NPOS / QB, NB);
  fa_readout_kernel<<<grid, 256, 0, stream>>>(mk, qk, mv, out);

  size_t memElems = (size_t)NB * CV_ * NPOS;   // 25,690,112
  int n4 = (int)(memElems / 4);
  qv_copy_kernel<<<(n4 + 255) / 256, 256, 0, stream>>>(
      (const float4*)qv, (float4*)(out + memElems), n4);
}
